// CrfRnn_74543452389982
// MI455X (gfx1250) — matrix operand order repaired and checked by reading, not compiled, not run on hardware
//
#include <hip/hip_runtime.h>
#include <stdint.h>

#define C 21
#define CP 11                // channel pairs (bf16 packed), pair 10 = {c20, 0}
#define HH 512
#define WW 512
#define NPIX (HH*WW)
#define RB 6
#define KB 13
#define RS 6
#define KS 13
#define TILE 16
#define HALO 6
#define LW (TILE + 2*HALO)   // 28
#define LA (LW*LW)           // 784

typedef __bf16 bf16v16 __attribute__((ext_vector_type(16)));
typedef float  f32v8   __attribute__((ext_vector_type(8)));
union BF16x16 { unsigned short u[16]; bf16v16 v; };

__device__ __forceinline__ unsigned short f2bf(float x) {
  unsigned int u = __float_as_uint(x);
  u += 0x7FFFu + ((u >> 16) & 1u);          // round-to-nearest-even
  return (unsigned short)(u >> 16);
}
__device__ __forceinline__ float bf2f(unsigned short h) {
  return __uint_as_float(((unsigned int)h) << 16);
}

// exp(-0.5*(d/3)^2) for d=-6..6
__constant__ float G13[13] = {
  0.13533528f, 0.24935221f, 0.41111229f, 0.60653066f, 0.80073740f,
  0.94595947f, 1.00000000f, 0.94595947f, 0.80073740f, 0.60653066f,
  0.41111229f, 0.24935221f, 0.13533528f };

// ---- prep: (H,W,C)->planar u/q, (H,W,3)->planar rgb --------------------
__global__ __launch_bounds__(256) void prep_kernel(
    const float* __restrict__ unaries, const float* __restrict__ rgb,
    float* __restrict__ u_pl, float* __restrict__ q, float* __restrict__ rgbp) {
  int i = blockIdx.x * 256 + threadIdx.x;
  #pragma unroll
  for (int c = 0; c < C; ++c) {
    float v = unaries[i * C + c];
    u_pl[c * NPIX + i] = v;
    q[c * NPIX + i]    = v;
  }
  #pragma unroll
  for (int ch = 0; ch < 3; ++ch) rgbp[ch * NPIX + i] = rgb[i * 3 + ch];
}

// ---- effective weights: A = -compat @ W, padded 32x32, bf16 ------------
__global__ __launch_bounds__(1024) void prepw_kernel(
    const float* __restrict__ wsp, const float* __restrict__ wbl,
    const float* __restrict__ compat,
    unsigned short* __restrict__ wspb, unsigned short* __restrict__ wblb) {
  int t = threadIdx.x;                    // 0..1023 = 32x32
  int r = t >> 5, k = t & 31;
  float vs = 0.f, vb = 0.f;
  if (r < C && k < C) {
    for (int m = 0; m < C; ++m) {
      float cm = compat[r * C + m];
      vs -= cm * wsp[m * C + k];
      vb -= cm * wbl[m * C + k];
    }
  }
  wspb[t] = f2bf(vs);
  wblb[t] = f2bf(vb);
}

__global__ __launch_bounds__(256) void zero_kernel(uint4* __restrict__ ptr) {
  ptr[blockIdx.x * 256 + threadIdx.x] = make_uint4(0u, 0u, 0u, 0u);
}

// ---- softmax over channels ---------------------------------------------
__global__ __launch_bounds__(256) void softmax_kernel(
    const float* __restrict__ q, float* __restrict__ p) {
  int i = blockIdx.x * 256 + threadIdx.x;
  float v[C];
  float m = -3.4e38f;
  #pragma unroll
  for (int c = 0; c < C; ++c) { v[c] = q[c * NPIX + i]; m = fmaxf(m, v[c]); }
  float s = 0.f;
  #pragma unroll
  for (int c = 0; c < C; ++c) { v[c] = __expf(v[c] - m); s += v[c]; }
  float inv = 1.0f / s;
  #pragma unroll
  for (int c = 0; c < C; ++c) p[c * NPIX + i] = v[c] * inv;
}

// ---- separable spatial blur: horizontal pass ---------------------------
__global__ __launch_bounds__(256) void spath_kernel(
    const float* __restrict__ p, float* __restrict__ tmp) {
  int i = blockIdx.x * 256 + threadIdx.x;
  int c = blockIdx.y;
  int x = i & (WW - 1);
  const float* row = p + (size_t)c * NPIX;
  float s = 0.f;
  #pragma unroll
  for (int k = 0; k < KS; ++k) {
    int xx = x + k - RS;
    if (xx >= 0 && xx < WW) s += G13[k] * row[i + k - RS];
  }
  tmp[(size_t)c * NPIX + i] = s;
}

// ---- vertical pass + border normalization + swizzled bf16 store --------
__global__ __launch_bounds__(256) void spatv_kernel(
    const float* __restrict__ tmp, unsigned short* __restrict__ spb) {
  int i = blockIdx.x * 256 + threadIdx.x;
  int c = blockIdx.y;
  int y = i >> 9, x = i & (WW - 1);
  const float* col = tmp + (size_t)c * NPIX;
  float s = 0.f, sx = 0.f, sy = 0.f;
  #pragma unroll
  for (int k = 0; k < KS; ++k) {
    int yy = y + k - RS;
    if (yy >= 0 && yy < HH) { s += G13[k] * col[i + (k - RS) * WW]; sy += G13[k]; }
    int xx = x + k - RS;
    if (xx >= 0 && xx < WW) sx += G13[k];
  }
  float val = s / (sx * sy);
  int t = i >> 4, n = i & 15;   // WMMA-B layout: lane = n + 16*(c>=16), elem = c&15
  spb[t * 512 + (n + ((c >> 4) << 4)) * 16 + (c & 15)] = f2bf(val);
}

// ---- bilateral 13x13, fused normalizer, packed-bf16-pair LDS -----------
__global__ __launch_bounds__(256) void bilateral_kernel(
    const float* __restrict__ p, const float* __restrict__ rgbp,
    unsigned short* __restrict__ blb) {
  __shared__ float4 rgl[LA];               // r,g,b,valid
  __shared__ unsigned int pl[CP * LA];     // 2 bf16 channels per dword
  int tid = threadIdx.x;
  int x0 = blockIdx.x * TILE, y0 = blockIdx.y * TILE;
  for (int idx = tid; idx < LA; idx += 256) {
    int ly = idx / LW, lx = idx - ly * LW;
    int gy = y0 + ly - HALO, gx = x0 + lx - HALO;
    bool ok = (gy >= 0) && (gy < HH) && (gx >= 0) && (gx < WW);
    int gi = ok ? (gy * WW + gx) : 0;
    float4 rv;
    rv.x = ok ? rgbp[0 * NPIX + gi] : 0.f;
    rv.y = ok ? rgbp[1 * NPIX + gi] : 0.f;
    rv.z = ok ? rgbp[2 * NPIX + gi] : 0.f;
    rv.w = ok ? 1.f : 0.f;
    rgl[idx] = rv;
    #pragma unroll
    for (int cp = 0; cp < CP; ++cp) {
      unsigned int lo = f2bf(p[(size_t)(2 * cp) * NPIX + gi]);
      unsigned int hi = (cp < CP - 1) ? (unsigned int)f2bf(p[(size_t)(2 * cp + 1) * NPIX + gi]) : 0u;
      pl[cp * LA + idx] = ok ? (lo | (hi << 16)) : 0u;
    }
  }
  __syncthreads();
  int ty = tid >> 4, tx = tid & 15;
  float4 cv = rgl[(ty + HALO) * LW + (tx + HALO)];
  float acc[C];
  #pragma unroll
  for (int c = 0; c < C; ++c) acc[c] = 0.f;
  float wsum = 0.f;
  const float h_a2 = 0.5f / 9.0f;   // 0.5/theta_alpha^2
  const float h_b2 = 0.5f / 9.0f;   // 0.5/theta_beta^2
  for (int dy = 0; dy < KB; ++dy) {
    int rowoff = (ty + dy) * LW + tx;
    float fy2 = (float)((dy - RB) * (dy - RB));
    #pragma unroll
    for (int dx = 0; dx < KB; ++dx) {
      int j = rowoff + dx;
      float4 rv = rgl[j];
      float dr = cv.x - rv.x, dg = cv.y - rv.y, db = cv.z - rv.z;
      float dcol = dr * dr + dg * dg + db * db;
      float dsp = fy2 + (float)((dx - RB) * (dx - RB));
      float w = __expf(-(dsp * h_a2 + dcol * h_b2)) * rv.w;   // valid masks OOB
      wsum += w;
      #pragma unroll
      for (int cp = 0; cp < CP - 1; ++cp) {
        unsigned int u = pl[cp * LA + j];
        acc[2 * cp]     += w * __uint_as_float(u << 16);
        acc[2 * cp + 1] += w * __uint_as_float(u & 0xFFFF0000u);
      }
      {
        unsigned int u = pl[(CP - 1) * LA + j];
        acc[20] += w * __uint_as_float(u << 16);
      }
    }
  }
  float inv = 1.0f / wsum;                 // wsum >= 1 (center tap)
  int i = (y0 + ty) * WW + (x0 + tx);
  int t = i >> 4, n = i & 15;
  #pragma unroll
  for (int c = 0; c < C; ++c)
    blb[t * 512 + (n + ((c >> 4) << 4)) * 16 + (c & 15)] = f2bf(acc[c] * inv);
}

// ---- WMMA channel mix: q = u + Asp@sp + Abl@bl -------------------------
__global__ __launch_bounds__(256) void gemm_kernel(
    const unsigned short* __restrict__ wspb, const unsigned short* __restrict__ wblb,
    const unsigned short* __restrict__ spb,  const unsigned short* __restrict__ blb,
    const float* __restrict__ u_pl, float* __restrict__ q) {
  int tid = threadIdx.x;
  int wave = tid >> 5, lane = tid & 31;
  int tile = blockIdx.x * 8 + wave;        // 16-pixel tile
  // A layout (16-bit 16x32): lane K = 2*kb + e (contiguous), kb=8*(lane>=16)
  int kb = (lane >> 4) << 3;
  int mr = lane & 15;
  BF16x16 a0s, a1s, a0b, a1b;
  #pragma unroll
  for (int e = 0; e < 16; ++e) {
    int K = 2 * kb + e;
    a0s.u[e] = wspb[mr * 32 + K];
    a1s.u[e] = wspb[(mr + 16) * 32 + K];
    a0b.u[e] = wblb[mr * 32 + K];
    a1b.u[e] = wblb[(mr + 16) * 32 + K];
  }
  // B pre-swizzled: 32 contiguous bytes per lane
  bf16v16 bs = *reinterpret_cast<const bf16v16*>(spb + (size_t)tile * 512 + lane * 16);
  bf16v16 bb = *reinterpret_cast<const bf16v16*>(blb + (size_t)tile * 512 + lane * 16);
  f32v8 acc0 = {}, acc1 = {};
  acc0 = __builtin_amdgcn_wmma_f32_16x16x32_bf16(false, a0s.v, false, bs, (short)0, acc0, false, false);
  acc0 = __builtin_amdgcn_wmma_f32_16x16x32_bf16(false, a0b.v, false, bb, (short)0, acc0, false, false);
  acc1 = __builtin_amdgcn_wmma_f32_16x16x32_bf16(false, a1s.v, false, bs, (short)0, acc1, false, false);
  acc1 = __builtin_amdgcn_wmma_f32_16x16x32_bf16(false, a1b.v, false, bb, (short)0, acc1, false, false);
  // D layout: vgpr r -> channel r+8*(lane>=16), pixel = lane&15
  int pix = tile * 16 + (lane & 15);
  int cb = (lane >> 4) << 3;
  #pragma unroll
  for (int r = 0; r < 8; ++r) {
    int c0 = cb + r;
    q[(size_t)c0 * NPIX + pix] = u_pl[(size_t)c0 * NPIX + pix] + acc0[r];
  }
  #pragma unroll
  for (int r = 0; r < 8; ++r) {
    int c1 = 16 + cb + r;
    if (c1 < C) q[(size_t)c1 * NPIX + pix] = u_pl[(size_t)c1 * NPIX + pix] + acc1[r];
  }
}

// ---- final: q[c][y*512+x] -> out[(x*512+y)*21+c] -----------------------
__global__ __launch_bounds__(256) void out_kernel(
    const float* __restrict__ q, float* __restrict__ out) {
  int i = blockIdx.x * 256 + threadIdx.x;
  int y = i >> 9, x = i & (WW - 1);
  int o = (x * HH + y) * C;
  #pragma unroll
  for (int c = 0; c < C; ++c) out[o + c] = q[(size_t)c * NPIX + i];
}

extern "C" void kernel_launch(void* const* d_in, const int* in_sizes, int n_in,
                              void* d_out, int out_size, void* d_ws, size_t ws_size,
                              hipStream_t stream) {
  (void)in_sizes; (void)n_in; (void)out_size; (void)ws_size;
  const float* unaries = (const float*)d_in[0];   // 512*512*21
  const float* rgb     = (const float*)d_in[1];   // 512*512*3
  const float* wsp     = (const float*)d_in[2];   // 21*21
  const float* wbl     = (const float*)d_in[3];   // 21*21
  const float* compat  = (const float*)d_in[4];   // 21*21
  float* outf = (float*)d_out;

  char* ws = (char*)d_ws;
  const size_t plane = (size_t)NPIX * 4;
  float* q    = (float*)(ws + 0  * plane);
  float* u_pl = (float*)(ws + 21 * plane);
  float* p    = (float*)(ws + 42 * plane);
  float* tmp  = (float*)(ws + 63 * plane);
  float* rgbp = (float*)(ws + 84 * plane);
  unsigned short* spb  = (unsigned short*)(ws + 87 * plane);
  unsigned short* blb  = (unsigned short*)(ws + 87 * plane + (size_t)32 * NPIX * 2);
  unsigned short* wspb = (unsigned short*)(ws + 87 * plane + (size_t)64 * NPIX * 2);
  unsigned short* wblb = wspb + 1024;

  prep_kernel<<<NPIX / 256, 256, 0, stream>>>(unaries, rgb, u_pl, q, rgbp);
  prepw_kernel<<<1, 1024, 0, stream>>>(wsp, wbl, compat, wspb, wblb);
  // clear both swizzled bf16 buffers (rows 21..31 must be zero for WMMA K=32)
  zero_kernel<<<((size_t)64 * NPIX * 2) / 16 / 256, 256, 0, stream>>>((uint4*)spb);

  for (int it = 0; it < 5; ++it) {
    softmax_kernel<<<NPIX / 256, 256, 0, stream>>>(q, p);
    spath_kernel<<<dim3(NPIX / 256, C), 256, 0, stream>>>(p, tmp);
    spatv_kernel<<<dim3(NPIX / 256, C), 256, 0, stream>>>(tmp, spb);
    bilateral_kernel<<<dim3(WW / TILE, HH / TILE), 256, 0, stream>>>(p, rgbp, blb);
    gemm_kernel<<<NPIX / 16 / 8, 256, 0, stream>>>(wspb, wblb, spb, blb, u_pl, q);
  }
  out_kernel<<<NPIX / 256, 256, 0, stream>>>(q, outf);
}
